// BasicTransformerBlock_72567767433935
// MI455X (gfx1250) — matrix operand order repaired and checked by reading, not compiled, not run on hardware
//
#include <hip/hip_runtime.h>
#include <hip/hip_bf16.h>

typedef __attribute__((ext_vector_type(16))) __bf16 v16bf;
typedef __attribute__((ext_vector_type(4)))  __bf16 v4bf;
typedef __attribute__((ext_vector_type(8)))  float  v8f;

#define HEADS 8
#define DIM_HEAD 64
#define TEXT_LEN 77
#define IMG_LEN 16
#define NQ 4096
#define BATCH 16
#define QD 320
#define CD 1024
#define IDIM 512
#define ATT_SCALE 0.125f   // 64^-0.5

static __device__ inline v8f wmma_bf16(v16bf a, v16bf b, v8f c) {
  return __builtin_amdgcn_wmma_f32_16x16x32_bf16(
      /*neg_a=*/false, a, /*neg_b=*/false, b,
      /*c_mod=*/(short)0, c, /*reuse_a=*/false, /*reuse_b=*/false);
}

// A-fragment (16x32 bf16): lane m=lane&15 holds row m; element i holds
// K = k0 + (i<8 ? 0 : 16) + half*8 + (i&7). Source row-major [row][ldk].
// Two contiguous 16B reads -> ds_load_b128 / global_load_b128.
static __device__ inline v16bf load_a_frag(const __bf16* base, int ldk, int row,
                                           int k0, int half) {
  v16bf a;
  const __bf16* p = base + (size_t)row * ldk + k0 + half * 8;
#pragma unroll
  for (int i = 0; i < 8; ++i) a[i] = p[i];
  p += 16;
#pragma unroll
  for (int i = 0; i < 8; ++i) a[8 + i] = p[i];
  return a;
}

// B-fragment (KxN, K=32): lane n=lane&15 holds column n; element i holds
// K = k0 + (i<8 ? 0 : 16) + half*8 + (i&7). Source holds B TRANSPOSED row-major [n][ldk] ->
// two contiguous 8-element runs -> two ds_load_b128.
static __device__ inline v16bf load_bt_frag(const __bf16* base, int ldk, int n,
                                            int k0, int half) {
  v16bf b;
  const __bf16* p = base + (size_t)n * ldk + k0 + half * 8;
#pragma unroll
  for (int i = 0; i < 8; ++i) b[i] = p[i];
  p += 16;
#pragma unroll
  for (int i = 0; i < 8; ++i) b[8 + i] = p[i];
  return b;
}

static __device__ inline v4bf cvt4(float4 f) {
  v4bf v;
  v[0] = (__bf16)f.x; v[1] = (__bf16)f.y; v[2] = (__bf16)f.z; v[3] = (__bf16)f.w;
  return v;
}

// ---------------------------------------------------------------------------
// GEMM: out_bf16[M, Ntot] = A_f32[rows mapped] @ W_f32[K, Ntot]
// Row mapping: r -> (r/batch_rows)*batch_stride + row_off + r%batch_rows
// 256 threads = 8 waves (4x2); tile 128x64, K-step 32; LDS double-buffered.
// GUARD=false for launches whose M is a multiple of 128 (no bounds checks).
// ---------------------------------------------------------------------------
template <bool GUARD>
__global__ __launch_bounds__(256) void gemm_proj(
    const float* __restrict__ A, int lda, int batch_rows, int batch_stride,
    int row_off, const float* __restrict__ W, int K, int Ntot, int M,
    __bf16* __restrict__ out) {
  __shared__ __bf16 smem[2 * (128 * 32 + 64 * 32)];  // 24 KB double buffer

  const int tid = threadIdx.x;
  const int bm = blockIdx.x, bn = blockIdx.y;
  const int wave = tid >> 5, lane = tid & 31;
  const int half = lane >> 4, m = lane & 15;
  const int wm = wave >> 1, wn = wave & 1;

  // hoisted per-thread staging pointers (row mapping done once)
  const float* aRow[4];
  bool aOk[4];
  int aIdx[4];
#pragma unroll
  for (int it = 0; it < 4; ++it) {
    int idx = tid * 4 + it * 1024;
    int r = idx >> 5, c = idx & 31;
    int gr = bm * 128 + r;
    aIdx[it] = idx;
    aOk[it] = !GUARD || (gr < M);
    int bi = gr / batch_rows;
    int ri = gr - bi * batch_rows;
    if (GUARD && gr >= M) { bi = 0; ri = 0; }
    aRow[it] = A + (size_t)(bi * batch_stride + row_off + ri) * lda + c;
  }
  const float* wPtr[2];
  int bC[2], bR[2];
#pragma unroll
  for (int it = 0; it < 2; ++it) {
    int idx = tid * 4 + it * 1024;
    int r = idx >> 6, c = idx & 63;
    bR[it] = r; bC[it] = c;
    wPtr[it] = W + (size_t)r * Ntot + bn * 64 + c;
  }

  v8f acc[2][2];
#pragma unroll
  for (int i = 0; i < 2; ++i)
#pragma unroll
    for (int j = 0; j < 2; ++j)
#pragma unroll
      for (int r = 0; r < 8; ++r) acc[i][j][r] = 0.f;

  const int nk = K >> 5;
  float4 fa[4], fb[2];

  // ---- tile loader (global -> regs) ----
  auto gload = [&](int kk) {
#pragma unroll
    for (int it = 0; it < 4; ++it) {
      if (aOk[it]) {
        fa[it] = *(const float4*)(aRow[it] + kk);
        __builtin_prefetch(aRow[it] + kk + 32, 0, 3);
      } else {
        fa[it] = make_float4(0.f, 0.f, 0.f, 0.f);
      }
    }
#pragma unroll
    for (int it = 0; it < 2; ++it)
      fb[it] = *(const float4*)(wPtr[it] + (size_t)kk * Ntot);
  };
  // ---- tile writer (regs -> LDS buffer) ----
  auto sstore = [&](int buf) {
    __bf16* sA  = smem + buf * (128 * 32 + 64 * 32);
    __bf16* sBt = sA + 128 * 32;
#pragma unroll
    for (int it = 0; it < 4; ++it) *(v4bf*)(sA + aIdx[it]) = cvt4(fa[it]);
#pragma unroll
    for (int it = 0; it < 2; ++it) {
      sBt[(bC[it] + 0) * 32 + bR[it]] = (__bf16)fb[it].x;
      sBt[(bC[it] + 1) * 32 + bR[it]] = (__bf16)fb[it].y;
      sBt[(bC[it] + 2) * 32 + bR[it]] = (__bf16)fb[it].z;
      sBt[(bC[it] + 3) * 32 + bR[it]] = (__bf16)fb[it].w;
    }
  };

  gload(0);
  sstore(0);
  __syncthreads();

  for (int t = 0; t < nk; ++t) {
    if (t + 1 < nk) gload((t + 1) * 32);   // long-latency loads issued first

    const __bf16* cA  = smem + (t & 1) * (128 * 32 + 64 * 32);
    const __bf16* cBt = cA + 128 * 32;
    v16bf a0 = load_a_frag(cA, 32, wm * 32 + m, 0, half);
    v16bf a1 = load_a_frag(cA, 32, wm * 32 + 16 + m, 0, half);
    v16bf b0 = load_bt_frag(cBt, 32, wn * 32 + m, 0, half);
    v16bf b1 = load_bt_frag(cBt, 32, wn * 32 + 16 + m, 0, half);
    acc[0][0] = wmma_bf16(a0, b0, acc[0][0]);
    acc[0][1] = wmma_bf16(a0, b1, acc[0][1]);
    acc[1][0] = wmma_bf16(a1, b0, acc[1][0]);
    acc[1][1] = wmma_bf16(a1, b1, acc[1][1]);

    if (t + 1 < nk) sstore((t + 1) & 1);
    __syncthreads();
  }

  // stage output tile in LDS (reuse smem), coalesced vector stores
  __bf16* sOut = smem;  // [128][64]
#pragma unroll
  for (int mt = 0; mt < 2; ++mt)
#pragma unroll
    for (int nt = 0; nt < 2; ++nt)
#pragma unroll
      for (int r = 0; r < 8; ++r)
        sOut[(wm * 32 + mt * 16 + half * 8 + r) * 64 + wn * 32 + nt * 16 + m] =
            (__bf16)acc[mt][nt][r];
  __syncthreads();

  {
    int lr = tid >> 1, seg = tid & 1;  // 128 rows x 2 segs of 32 cols
    int row = bm * 128 + lr;
    if (!GUARD || row < M) {
      const uint4* src = (const uint4*)(sOut + lr * 64 + seg * 32);
      uint4* dst = (uint4*)(out + (size_t)row * Ntot + bn * 64 + seg * 32);
#pragma unroll
      for (int i = 0; i < 4; ++i) dst[i] = src[i];
    }
  }
}

// ---------------------------------------------------------------------------
// Final GEMM: out_f32[M,320] = merged_bf16[M,512] @ Wo_f32[512,320] + bo
// M is a multiple of 128: no bounds checks. LDS double-buffered.
// ---------------------------------------------------------------------------
__global__ __launch_bounds__(256) void gemm_out(
    const __bf16* __restrict__ A, const float* __restrict__ W,
    const float* __restrict__ bias, float* __restrict__ out, int M, int K,
    int Ntot) {
  __shared__ float smemf[128 * 64];  // 32 KB: aliased double buffer + f32 out
  __bf16* smem = (__bf16*)smemf;     // 16384 bf16 slots >= 2*6144

  const int tid = threadIdx.x;
  const int bm = blockIdx.x, bn = blockIdx.y;
  const int wave = tid >> 5, lane = tid & 31;
  const int half = lane >> 4, m = lane & 15;
  const int wm = wave >> 1, wn = wave & 1;

  const __bf16* aRow[4];
  int aIdx[4];
#pragma unroll
  for (int it = 0; it < 4; ++it) {
    int idx = tid * 4 + it * 1024;
    int r = idx >> 5, c = idx & 31;
    aIdx[it] = idx;
    aRow[it] = A + (size_t)(bm * 128 + r) * K + c;
  }
  const float* wPtr[2];
  int bC[2], bR[2];
#pragma unroll
  for (int it = 0; it < 2; ++it) {
    int idx = tid * 4 + it * 1024;
    int r = idx >> 6, c = idx & 63;
    bR[it] = r; bC[it] = c;
    wPtr[it] = W + (size_t)r * Ntot + bn * 64 + c;
  }

  v8f acc[2][2];
#pragma unroll
  for (int i = 0; i < 2; ++i)
#pragma unroll
    for (int j = 0; j < 2; ++j)
#pragma unroll
      for (int r = 0; r < 8; ++r) acc[i][j][r] = 0.f;

  const int nk = K >> 5;
  v4bf va[4];
  float4 fb[2];

  auto gload = [&](int kk) {
#pragma unroll
    for (int it = 0; it < 4; ++it) {
      va[it] = *(const v4bf*)(aRow[it] + kk);
      __builtin_prefetch(aRow[it] + kk + 32, 0, 3);
    }
#pragma unroll
    for (int it = 0; it < 2; ++it)
      fb[it] = *(const float4*)(wPtr[it] + (size_t)kk * Ntot);
  };
  auto sstore = [&](int buf) {
    __bf16* sA  = smem + buf * (128 * 32 + 64 * 32);
    __bf16* sBt = sA + 128 * 32;
#pragma unroll
    for (int it = 0; it < 4; ++it) *(v4bf*)(sA + aIdx[it]) = va[it];
#pragma unroll
    for (int it = 0; it < 2; ++it) {
      sBt[(bC[it] + 0) * 32 + bR[it]] = (__bf16)fb[it].x;
      sBt[(bC[it] + 1) * 32 + bR[it]] = (__bf16)fb[it].y;
      sBt[(bC[it] + 2) * 32 + bR[it]] = (__bf16)fb[it].z;
      sBt[(bC[it] + 3) * 32 + bR[it]] = (__bf16)fb[it].w;
    }
  };

  gload(0);
  sstore(0);
  __syncthreads();

  for (int t = 0; t < nk; ++t) {
    if (t + 1 < nk) gload((t + 1) * 32);

    const __bf16* cA  = smem + (t & 1) * (128 * 32 + 64 * 32);
    const __bf16* cBt = cA + 128 * 32;
    v16bf a0 = load_a_frag(cA, 32, wm * 32 + m, 0, half);
    v16bf a1 = load_a_frag(cA, 32, wm * 32 + 16 + m, 0, half);
    v16bf b0 = load_bt_frag(cBt, 32, wn * 32 + m, 0, half);
    v16bf b1 = load_bt_frag(cBt, 32, wn * 32 + 16 + m, 0, half);
    acc[0][0] = wmma_bf16(a0, b0, acc[0][0]);
    acc[0][1] = wmma_bf16(a0, b1, acc[0][1]);
    acc[1][0] = wmma_bf16(a1, b0, acc[1][0]);
    acc[1][1] = wmma_bf16(a1, b1, acc[1][1]);

    if (t + 1 < nk) sstore((t + 1) & 1);
    __syncthreads();
  }

  // bias + f32 staging + coalesced stores (always full tile)
  float b0c = bias[bn * 64 + wn * 32 + m];
  float b1c = bias[bn * 64 + wn * 32 + 16 + m];
#pragma unroll
  for (int mt = 0; mt < 2; ++mt)
#pragma unroll
    for (int r = 0; r < 8; ++r) {
      smemf[(wm * 32 + mt * 16 + half * 8 + r) * 64 + wn * 32 + m] =
          acc[mt][0][r] + b0c;
      smemf[(wm * 32 + mt * 16 + half * 8 + r) * 64 + wn * 32 + 16 + m] =
          acc[mt][1][r] + b1c;
    }
  __syncthreads();

  {
    int lr = tid >> 1, seg = tid & 1;  // 32 f32 per thread
    int row = bm * 128 + lr;
    const uint4* src = (const uint4*)(smemf + lr * 64 + seg * 32);
    uint4* dst = (uint4*)(out + (size_t)row * Ntot + bn * 64 + seg * 32);
#pragma unroll
    for (int i = 0; i < 8; ++i) dst[i] = src[i];
  }
}

// ---------------------------------------------------------------------------
// Fused dual attention (text 77 keys + image 16 keys), per (b,h,128 q rows).
// 8 waves x 16 query rows. K row-major [j][d] (Q@K^T B-frags contiguous);
// V transposed [d][j] so P@V B-frags contiguous too.
// ---------------------------------------------------------------------------
__global__ __launch_bounds__(256) void attn_kernel(
    const __bf16* __restrict__ qws, const __bf16* __restrict__ kt,
    const __bf16* __restrict__ vt, const __bf16* __restrict__ ki,
    const __bf16* __restrict__ vi, __bf16* __restrict__ merged,
    const float* __restrict__ tscale, const float* __restrict__ iscale) {
  __shared__ __bf16 sK[80 * 64];      // text keys [j][d], rows >=77 zero
  __shared__ __bf16 sVt[64 * 96];     // text V transposed [d][j], j>=77 zero
  __shared__ __bf16 sKi[16 * 64];     // image keys [j][d]
  __shared__ __bf16 sVit[64 * 32];    // image V transposed [d][j], j>=16 zero
  __shared__ __bf16 sP[8 * 16 * 96];  // per-wave P / output staging

  const int bh = blockIdx.x;
  const int b = bh >> 3, h = bh & 7;
  const int rowblk = blockIdx.y;
  const int tid = threadIdx.x;

  for (int idx = tid; idx < 80 * 64; idx += 256) {
    int j = idx >> 6, d = idx & 63;
    sK[idx] = (j < TEXT_LEN)
                  ? kt[(size_t)(b * TEXT_LEN + j) * IDIM + h * 64 + d]
                  : (__bf16)0.f;
  }
  for (int idx = tid; idx < 96 * 64; idx += 256) {
    int j = idx >> 6, d = idx & 63;  // coalesced read, transposed write
    __bf16 v = (j < TEXT_LEN)
                   ? vt[(size_t)(b * TEXT_LEN + j) * IDIM + h * 64 + d]
                   : (__bf16)0.f;
    sVt[d * 96 + j] = v;
  }
  for (int idx = tid; idx < 16 * 64; idx += 256) {
    int j = idx >> 6, d = idx & 63;
    sKi[idx] = ki[(size_t)(b * IMG_LEN + j) * IDIM + h * 64 + d];
  }
  for (int idx = tid; idx < 32 * 64; idx += 256) {
    int j = idx >> 6, d = idx & 63;
    __bf16 v = (j < IMG_LEN)
                   ? vi[(size_t)(b * IMG_LEN + j) * IDIM + h * 64 + d]
                   : (__bf16)0.f;
    sVit[d * 32 + j] = v;
  }
  __syncthreads();

  const int wave = tid >> 5, lane = tid & 31;
  const int half = lane >> 4, m = lane & 15;
  const int qrow0 = rowblk * 128 + wave * 16;
  const __bf16* qptr = qws + (size_t)(b * NQ + qrow0) * IDIM + h * 64;
  __bf16* myP = sP + wave * 16 * 96;

  v16bf aq0 = load_a_frag(qptr, IDIM, m, 0, half);
  v16bf aq1 = load_a_frag(qptr, IDIM, m, 32, half);

  // ---- text: S = Q K^T ----------------------------------------------------
  v8f st[5];
#pragma unroll
  for (int nt = 0; nt < 5; ++nt) {
#pragma unroll
    for (int r = 0; r < 8; ++r) st[nt][r] = 0.f;
    st[nt] = wmma_bf16(aq0, load_bt_frag(sK, 64, nt * 16 + m, 0, half), st[nt]);
    st[nt] = wmma_bf16(aq1, load_bt_frag(sK, 64, nt * 16 + m, 32, half), st[nt]);
  }
#pragma unroll
  for (int nt = 0; nt < 5; ++nt) {
    int n = nt * 16 + m;
#pragma unroll
    for (int r = 0; r < 8; ++r) {
      float s = st[nt][r] * ATT_SCALE;
      st[nt][r] = (n < TEXT_LEN) ? s : -1e30f;
    }
  }
  float rmax[8], rsum[8];
#pragma unroll
  for (int r = 0; r < 8; ++r) {
    float mx = st[0][r];
#pragma unroll
    for (int nt = 1; nt < 5; ++nt) mx = fmaxf(mx, st[nt][r]);
#pragma unroll
    for (int off = 1; off < 16; off <<= 1)
      mx = fmaxf(mx, __shfl_xor(mx, off, 32));
    rmax[r] = mx;
  }
#pragma unroll
  for (int r = 0; r < 8; ++r) {
    float sum = 0.f;
#pragma unroll
    for (int nt = 0; nt < 5; ++nt) {
      float p = __expf(st[nt][r] - rmax[r]);
      st[nt][r] = p;
      sum += p;
    }
#pragma unroll
    for (int off = 1; off < 16; off <<= 1) sum += __shfl_xor(sum, off, 32);
    rsum[r] = 1.f / sum;
  }
#pragma unroll
  for (int i = 0; i < 8; ++i) myP[m * 96 + 80 + half * 8 + i] = (__bf16)0.f;
#pragma unroll
  for (int nt = 0; nt < 5; ++nt)
#pragma unroll
    for (int r = 0; r < 8; ++r)
      myP[(half * 8 + r) * 96 + nt * 16 + m] = (__bf16)(st[nt][r] * rsum[r]);
  __syncthreads();

  v8f ot[4];
#pragma unroll
  for (int nt = 0; nt < 4; ++nt)
#pragma unroll
    for (int r = 0; r < 8; ++r) ot[nt][r] = 0.f;
#pragma unroll
  for (int ks = 0; ks < 3; ++ks) {
    v16bf ap = load_a_frag(myP, 96, m, ks * 32, half);
#pragma unroll
    for (int nt = 0; nt < 4; ++nt)
      ot[nt] = wmma_bf16(
          ap, load_bt_frag(sVt, 96, nt * 16 + m, ks * 32, half), ot[nt]);
  }

  // ---- image: 16 keys -----------------------------------------------------
  v8f si;
#pragma unroll
  for (int r = 0; r < 8; ++r) si[r] = 0.f;
  si = wmma_bf16(aq0, load_bt_frag(sKi, 64, m, 0, half), si);
  si = wmma_bf16(aq1, load_bt_frag(sKi, 64, m, 32, half), si);

  float isum[8];
#pragma unroll
  for (int r = 0; r < 8; ++r) {
    float s = si[r] * ATT_SCALE;
    float mx = s;
#pragma unroll
    for (int off = 1; off < 16; off <<= 1)
      mx = fmaxf(mx, __shfl_xor(mx, off, 32));
    float p = __expf(s - mx);
    si[r] = p;
    float sum = p;
#pragma unroll
    for (int off = 1; off < 16; off <<= 1) sum += __shfl_xor(sum, off, 32);
    isum[r] = 1.f / sum;
  }
  __syncthreads();  // all waves done reading P before overwrite
#pragma unroll
  for (int i = 0; i < 8; ++i) myP[m * 96 + 16 + half * 8 + i] = (__bf16)0.f;
#pragma unroll
  for (int r = 0; r < 8; ++r)
    myP[(half * 8 + r) * 96 + m] = (__bf16)(si[r] * isum[r]);
  __syncthreads();

  v8f oi[4];
#pragma unroll
  for (int nt = 0; nt < 4; ++nt)
#pragma unroll
    for (int r = 0; r < 8; ++r) oi[nt][r] = 0.f;
  v16bf api = load_a_frag(myP, 96, m, 0, half);
#pragma unroll
  for (int nt = 0; nt < 4; ++nt)
    oi[nt] = wmma_bf16(api, load_bt_frag(sVit, 32, nt * 16 + m, 0, half),
                       oi[nt]);

  // combine, stage 16x64 tile, coalesced vector stores
  const float ts = tscale[0], is = iscale[0];
  __syncthreads();
#pragma unroll
  for (int nt = 0; nt < 4; ++nt)
#pragma unroll
    for (int r = 0; r < 8; ++r)
      myP[(half * 8 + r) * 96 + nt * 16 + m] =
          (__bf16)(ts * ot[nt][r] + is * oi[nt][r]);
  __syncthreads();

  {
    int lr = lane >> 1, seg = lane & 1;  // 16 rows x 2 segs of 32 cols
    int row = b * NQ + qrow0 + lr;
    const uint4* src = (const uint4*)(myP + lr * 96 + seg * 32);
    uint4* dst = (uint4*)(merged + (size_t)row * IDIM + h * 64 + seg * 32);
#pragma unroll
    for (int i = 0; i < 4; ++i) dst[i] = src[i];
  }
}

// ---------------------------------------------------------------------------
extern "C" void kernel_launch(void* const* d_in, const int* in_sizes, int n_in,
                              void* d_out, int out_size, void* d_ws,
                              size_t ws_size, hipStream_t stream) {
  const float* x       = (const float*)d_in[0];
  const float* context = (const float*)d_in[1];
  const float* Wq      = (const float*)d_in[2];
  const float* Wk      = (const float*)d_in[3];
  const float* Wv      = (const float*)d_in[4];
  const float* Wk_ip   = (const float*)d_in[5];
  const float* Wv_ip   = (const float*)d_in[6];
  const float* Wo      = (const float*)d_in[7];
  const float* bo      = (const float*)d_in[8];
  const float* tscale  = (const float*)d_in[9];
  const float* iscale  = (const float*)d_in[10];
  float* out = (float*)d_out;

  const int M = BATCH * NQ;  // 65536
  char* ws = (char*)d_ws;
  size_t off = 0;
  __bf16* q_ws = (__bf16*)(ws + off);      off += (size_t)M * IDIM * 2;
  __bf16* merged = (__bf16*)(ws + off);    off += (size_t)M * IDIM * 2;
  __bf16* kt_ws = (__bf16*)(ws + off);     off += (size_t)BATCH * TEXT_LEN * IDIM * 2;
  __bf16* vt_ws = (__bf16*)(ws + off);     off += (size_t)BATCH * TEXT_LEN * IDIM * 2;
  __bf16* ki_ws = (__bf16*)(ws + off);     off += (size_t)BATCH * IMG_LEN * IDIM * 2;
  __bf16* vi_ws = (__bf16*)(ws + off);     off += (size_t)BATCH * IMG_LEN * IDIM * 2;

  // q = x @ Wq  (65536 x 320 x 512) -- full tiles, no guards
  gemm_proj<false><<<dim3(M / 128, IDIM / 64), 256, 0, stream>>>(
      x, QD, NQ, NQ, 0, Wq, QD, IDIM, M, q_ws);

  // text k/v: rows = first 77 of each 93-row context block (partial last tile)
  const int Mt = BATCH * TEXT_LEN;  // 1232
  gemm_proj<true><<<dim3((Mt + 127) / 128, IDIM / 64), 256, 0, stream>>>(
      context, CD, TEXT_LEN, 93, 0, Wk, CD, IDIM, Mt, kt_ws);
  gemm_proj<true><<<dim3((Mt + 127) / 128, IDIM / 64), 256, 0, stream>>>(
      context, CD, TEXT_LEN, 93, 0, Wv, CD, IDIM, Mt, vt_ws);

  // image k/v: rows 77..92 of each context block (256 rows -> full tiles)
  const int Mi = BATCH * IMG_LEN;  // 256
  gemm_proj<false><<<dim3(Mi / 128, IDIM / 64), 256, 0, stream>>>(
      context, CD, IMG_LEN, 93, TEXT_LEN, Wk_ip, CD, IDIM, Mi, ki_ws);
  gemm_proj<false><<<dim3(Mi / 128, IDIM / 64), 256, 0, stream>>>(
      context, CD, IMG_LEN, 93, TEXT_LEN, Wv_ip, CD, IDIM, Mi, vi_ws);

  // fused dual attention
  attn_kernel<<<dim3(BATCH * HEADS, NQ / 128), 256, 0, stream>>>(
      q_ws, kt_ws, vt_ws, ki_ws, vi_ws, merged, tscale, iscale);

  // out = merged @ Wo + bo  (65536 x 512 x 320) -- full tiles
  gemm_out<<<dim3(M / 128, QD / 64), 256, 0, stream>>>(
      merged, Wo, bo, out, M, IDIM, QD);
}
